// PDFC_28200755266008
// MI455X (gfx1250) — matrix operand order repaired and checked by reading, not compiled, not run on hardware
//
#include <hip/hip_runtime.h>
#include <hip/hip_bf16.h>
#include <math.h>

// ---- problem constants fixed by the reference ----
#define HD   256   // H*D
#define HH   4     // heads
#define DD   64    // per-head dim
#define OUTF 64    // output feature dim
#define KMAX 256   // max GEMM K (F1 = SIN = HD = 256)

typedef __attribute__((ext_vector_type(16))) __bf16 v16bf;
typedef __attribute__((ext_vector_type(8)))  float  v8f;

__device__ __forceinline__ float waveSum(float v) {
#pragma unroll
  for (int o = 16; o; o >>= 1) v += __shfl_xor(v, o, 32);
  return v;
}

__device__ __forceinline__ void atomicMaxF(float* a, float v) {
  if (v >= 0.f) atomicMax((int*)a, __float_as_int(v));
  else          atomicMin((unsigned int*)a, __float_as_uint(v));
}

// ---------------- generic fill (grid-stride) ----------------
__global__ void k_fill(float* __restrict__ p, size_t n, float v) {
  size_t i = (size_t)blockIdx.x * blockDim.x + threadIdx.x;
  size_t st = (size_t)gridDim.x * blockDim.x;
  for (; i < n; i += st) p[i] = v;
}

// ---------------- WMMA bf16 GEMM: C = A(MxK) @ B(KxN) (+bias) ----------------
// Block = 256 threads = 8 waves. blockIdx.y selects one 16-col N tile; the
// K x 16 weight panel is staged once into LDS, pre-swizzled into WMMA
// B-fragment order (packed bf16), so each lane's B read per K-step is one
// contiguous 32B LDS access (2 x ds_load_b128). Each wave computes two 16x16
// M tiles sharing that B fragment -> 2 back-to-back v_wmma per K-step.
// Fragment layouts per CDNA5 ISA 7.12.2 (wave32).
__global__ void __launch_bounds__(256)
k_gemm(const float* __restrict__ A, const float* __restrict__ B,
       const float* __restrict__ bias, float* __restrict__ C,
       int M, int Ncol, int K) {
  __shared__ __align__(32) unsigned int ldsB[(KMAX / 32) * 256];

  const int tilesM = M >> 4;
  const int nkb = K >> 5;

  // ---- stage B panel into LDS in fragment order ----
  // thread t = (flane, fd): dword fd of lane flane holds B[k0+8*fh+2*fd+8*(fd>=4)+{0,1}, col]
  {
    const int t = threadIdx.x;
    const int flane = t >> 3, fd = t & 7;
    const int fh = flane >> 4, flr = flane & 15;
    const int bcolF = blockIdx.y * 16 + flr;
    for (int kb = 0; kb < nkb; ++kb) {
      int krow = kb * 32 + 8 * fh + 2 * fd + (fd >= 4 ? 8 : 0);
      float x0 = B[(size_t)krow * Ncol + bcolF];
      float x1 = B[(size_t)(krow + 1) * Ncol + bcolF];
      union { __bf16 h[2]; unsigned int u; } p;
      p.h[0] = (__bf16)x0; p.h[1] = (__bf16)x1;
      ldsB[kb * 256 + flane * 8 + fd] = p.u;
    }
  }
  __syncthreads();

  const int wv   = threadIdx.x >> 5;
  const int lane = threadIdx.x & 31;
  const int half = lane >> 4, lr = lane & 15;
  const int tm0 = blockIdx.x * 16 + wv * 2;   // two consecutive M tiles per wave
  const int tm1 = tm0 + 1;
  const bool act0 = tm0 < tilesM;
  const bool act1 = tm1 < tilesM;
  if (!act0) return;                           // wave-uniform; EXEC stays all-1
  const int bcol = blockIdx.y * 16 + lr;
  const int arow0 = tm0 * 16 + lr;
  const int arow1 = tm1 * 16 + lr;

  v8f acc0 = {0.f, 0.f, 0.f, 0.f, 0.f, 0.f, 0.f, 0.f};
  v8f acc1 = {0.f, 0.f, 0.f, 0.f, 0.f, 0.f, 0.f, 0.f};

  for (int k0 = 0; k0 < K; k0 += 32) {
    // B fragment: 32 contiguous bytes per lane from LDS
    v16bf b = *(const v16bf*)&ldsB[(k0 >> 5) * 256 + lane * 8];

    // A fragment tile 0: K = k0 + 8*half + {0..7} and + 16 more
    {
      const float* Ap = A + (size_t)arow0 * K + k0 + 8 * half;
      float4 xa = *(const float4*)(Ap);
      float4 xb = *(const float4*)(Ap + 4);
      float4 ya = *(const float4*)(Ap + 16);
      float4 yb = *(const float4*)(Ap + 20);
      v16bf a;
      a[0]  = (__bf16)xa.x; a[1]  = (__bf16)xa.y; a[2]  = (__bf16)xa.z; a[3]  = (__bf16)xa.w;
      a[4]  = (__bf16)xb.x; a[5]  = (__bf16)xb.y; a[6]  = (__bf16)xb.z; a[7]  = (__bf16)xb.w;
      a[8]  = (__bf16)ya.x; a[9]  = (__bf16)ya.y; a[10] = (__bf16)ya.z; a[11] = (__bf16)ya.w;
      a[12] = (__bf16)yb.x; a[13] = (__bf16)yb.y; a[14] = (__bf16)yb.z; a[15] = (__bf16)yb.w;
      acc0 = __builtin_amdgcn_wmma_f32_16x16x32_bf16(false, a, false, b,
                                                     (short)0, acc0, false, false);
    }
    if (act1) {
      const float* Ap = A + (size_t)arow1 * K + k0 + 8 * half;
      float4 xa = *(const float4*)(Ap);
      float4 xb = *(const float4*)(Ap + 4);
      float4 ya = *(const float4*)(Ap + 16);
      float4 yb = *(const float4*)(Ap + 20);
      v16bf a;
      a[0]  = (__bf16)xa.x; a[1]  = (__bf16)xa.y; a[2]  = (__bf16)xa.z; a[3]  = (__bf16)xa.w;
      a[4]  = (__bf16)xb.x; a[5]  = (__bf16)xb.y; a[6]  = (__bf16)xb.z; a[7]  = (__bf16)xb.w;
      a[8]  = (__bf16)ya.x; a[9]  = (__bf16)ya.y; a[10] = (__bf16)ya.z; a[11] = (__bf16)ya.w;
      a[12] = (__bf16)yb.x; a[13] = (__bf16)yb.y; a[14] = (__bf16)yb.z; a[15] = (__bf16)yb.w;
      acc1 = __builtin_amdgcn_wmma_f32_16x16x32_bf16(false, a, false, b,
                                                     (short)0, acc1, false, false);
    }
  }

  const float bv = bias ? bias[bcol] : 0.f;
#pragma unroll
  for (int v = 0; v < 8; ++v) {                // D layout: M = v + 8*half
    int r0 = tm0 * 16 + 8 * half + v;
    C[(size_t)r0 * Ncol + bcol] = acc0[v] + bv;
  }
  if (act1) {
#pragma unroll
    for (int v = 0; v < 8; ++v) {
      int r1 = tm1 * 16 + 8 * half + v;
      C[(size_t)r1 * Ncol + bcol] = acc1[v] + bv;
    }
  }
}

// ---------------- GAT: el/er per (node, head), one wave each ----------------
__global__ void k_elr(const float* __restrict__ h, const float* __restrict__ al,
                      const float* __restrict__ ar, float* __restrict__ el,
                      float* __restrict__ er, int N) {
  int w = blockIdx.x * (blockDim.x >> 5) + (threadIdx.x >> 5);
  if (w >= N * HH) return;
  int lane = threadIdx.x & 31;
  int n = w >> 2, hd = w & 3;
  const float* hp  = h + (size_t)n * HD + hd * DD;
  const float* alp = al + hd * DD;
  const float* arp = ar + hd * DD;
  float h1v = hp[lane], h2v = hp[lane + 32];
  float e1 = waveSum(h1v * alp[lane] + h2v * alp[lane + 32]);
  float e2 = waveSum(h1v * arp[lane] + h2v * arp[lane + 32]);
  if (lane == 0) { el[w] = e1; er[w] = e2; }
}

__global__ void k_att_max(const int* __restrict__ src, const int* __restrict__ dst,
                          const float* __restrict__ el, const float* __restrict__ er,
                          float* __restrict__ m, int E) {
  int i = blockIdx.x * blockDim.x + threadIdx.x;
  if (i >= E * HH) return;
  int e = i >> 2, hd = i & 3;
  int d = dst[e];
  float x = el[src[e] * HH + hd] + er[d * HH + hd];
  x = x > 0.f ? x : 0.2f * x;
  atomicMaxF(&m[d * HH + hd], x);
}

__global__ void k_att_ps(const int* __restrict__ src, const int* __restrict__ dst,
                         const float* __restrict__ el, const float* __restrict__ er,
                         const float* __restrict__ m, float* __restrict__ s,
                         float* __restrict__ a, int E) {
  int i = blockIdx.x * blockDim.x + threadIdx.x;
  if (i >= E * HH) return;
  int e = i >> 2, hd = i & 3;
  int d = dst[e];
  float x = el[src[e] * HH + hd] + er[d * HH + hd];
  x = x > 0.f ? x : 0.2f * x;
  float p = __expf(x - m[d * HH + hd]);
  a[i] = p;
  atomicAdd(&s[d * HH + hd], p);
}

__global__ void k_att_norm(const int* __restrict__ dst, const float* __restrict__ s,
                           float* __restrict__ a, int E) {
  int i = blockIdx.x * blockDim.x + threadIdx.x;
  if (i >= E * HH) return;
  int e = i >> 2, hd = i & 3;
  a[i] /= s[dst[e] * HH + hd];
}

// one block (256 threads) per edge: coalesced gather of h[src], atomic scatter to out[dst]
__global__ void k_att_scatter(const int* __restrict__ src, const int* __restrict__ dst,
                              const float* __restrict__ a, const float* __restrict__ h,
                              float* __restrict__ out, int E) {
  int e = blockIdx.x, c = threadIdx.x;
  float av = a[(size_t)e * HH + (c >> 6)];
  atomicAdd(&out[(size_t)dst[e] * HD + c], av * h[(size_t)src[e] * HD + c]);
}

__global__ void k_biasclip(float* __restrict__ z, const float* __restrict__ b, size_t n) {
  size_t i = (size_t)blockIdx.x * blockDim.x + threadIdx.x;
  if (i >= n) return;
  float v = z[i] + b[i & 255];               // HD == 256
  z[i] = fminf(fmaxf(v, 0.f), 6.f);
}

// ---------------- semantic attention ----------------
__global__ void k_tanhdot(const float* __restrict__ T, const float* __restrict__ wp2,
                          float* __restrict__ wsum, int N) {
  int w = blockIdx.x * (blockDim.x >> 5) + (threadIdx.x >> 5);
  if (w >= N) return;
  int lane = threadIdx.x & 31;
  const float* t = T + (size_t)w * OUTF;
  float v = tanhf(t[lane]) * wp2[lane] + tanhf(t[lane + 32]) * wp2[lane + 32];
  v = waveSum(v);
  if (lane == 0) atomicAdd(wsum, v);
}

__global__ void k_beta(const float* __restrict__ wsum, float* __restrict__ beta, float invN) {
  float w0 = wsum[0] * invN, w1 = wsum[1] * invN;
  float mx = fmaxf(w0, w1);
  float e0 = __expf(w0 - mx), e1 = __expf(w1 - mx);
  float s = e0 + e1;
  beta[0] = e0 / s; beta[1] = e1 / s;
}

__global__ void k_fuse(const float* __restrict__ z0, const float* __restrict__ z1,
                       const float* __restrict__ beta, float* __restrict__ f, size_t n) {
  size_t i = (size_t)blockIdx.x * blockDim.x + threadIdx.x;
  if (i < n) f[i] = beta[0] * z0[i] + beta[1] * z1[i];
}

// ---------------- GCN ----------------
__global__ void k_deg(const int* __restrict__ gs, const int* __restrict__ gd,
                      float* __restrict__ dego, float* __restrict__ degi, int E) {
  int e = blockIdx.x * blockDim.x + threadIdx.x;
  if (e >= E) return;
  atomicAdd(&dego[gs[e]], 1.f);
  atomicAdd(&degi[gd[e]], 1.f);
}

__global__ void k_rsq(const float* __restrict__ d, float* __restrict__ o, int N) {
  int i = blockIdx.x * blockDim.x + threadIdx.x;
  if (i < N) o[i] = rsqrtf(fmaxf(d[i], 1.f));
}

__global__ void k_gcn_scatter(const int* __restrict__ gs, const int* __restrict__ gd,
                              const float* __restrict__ xw, const float* __restrict__ dvo,
                              float* __restrict__ acc, int E) {
  int idx = blockIdx.x * blockDim.x + threadIdx.x;
  if (idx >= E * OUTF) return;
  int e = idx >> 6, c = idx & 63;
  int s = gs[e];
  atomicAdd(&acc[(size_t)gd[e] * OUTF + c], xw[(size_t)s * OUTF + c] * dvo[s]);
}

__global__ void k_gcnfin(const float* __restrict__ acc, const float* __restrict__ dvi,
                         const float* __restrict__ b, float* __restrict__ out,
                         int N, int dotanh) {
  int i = blockIdx.x * blockDim.x + threadIdx.x;
  if (i >= N * OUTF) return;
  int n = i >> 6, c = i & 63;
  float v = acc[i] * dvi[n];
  if (b) v += b[c];
  out[i] = dotanh ? tanhf(v) : v;
}

__global__ void k_zout(const float* __restrict__ Fe, const float* __restrict__ P,
                       float* __restrict__ Z, size_t n) {
  size_t i = (size_t)blockIdx.x * blockDim.x + threadIdx.x;
  if (i < n) Z[i] = 0.5f * (Fe[i] + P[i]);
}

__global__ void k_tr(const int* __restrict__ as, const int* __restrict__ ad,
                     const float* __restrict__ Z, const int* __restrict__ kp,
                     float* __restrict__ tr, int E) {
  int kk = *kp;
  float acc = 0.f;
  for (int e = blockIdx.x * blockDim.x + threadIdx.x; e < E;
       e += gridDim.x * blockDim.x) {
    const float* a = Z + (size_t)as[e] * OUTF;
    const float* b = Z + (size_t)ad[e] * OUTF;
    for (int c = 0; c < kk; ++c) { float d = a[c] - b[c]; acc += d * d; }
  }
  acc = waveSum(acc);
  if ((threadIdx.x & 31) == 0) atomicAdd(tr, 0.5f * acc);
}

// =====================================================================
extern "C" void kernel_launch(void* const* d_in, const int* in_sizes, int n_in,
                              void* d_out, int out_size, void* d_ws, size_t ws_size,
                              hipStream_t stream) {
  (void)n_in; (void)out_size; (void)ws_size;
  const float* feat0 = (const float*)d_in[0];
  const float* feat1 = (const float*)d_in[1];
  const float* sen   = (const float*)d_in[2];
  const float* Wgat0 = (const float*)d_in[3];
  const float* al0   = (const float*)d_in[4];
  const float* ar0   = (const float*)d_in[5];
  const float* b0    = (const float*)d_in[6];
  const float* Wgat1 = (const float*)d_in[7];
  const float* al1   = (const float*)d_in[8];
  const float* ar1   = (const float*)d_in[9];
  const float* b1    = (const float*)d_in[10];
  const float* Wp1   = (const float*)d_in[11];
  const float* bp1   = (const float*)d_in[12];
  const float* wp2   = (const float*)d_in[13];
  const float* Wf    = (const float*)d_in[14];
  const float* bf    = (const float*)d_in[15];
  const float* Wg1   = (const float*)d_in[16];
  const float* bg1   = (const float*)d_in[17];
  const float* Wg2   = (const float*)d_in[18];
  const int* src0 = (const int*)d_in[19];
  const int* dst0 = (const int*)d_in[20];
  const int* src1 = (const int*)d_in[21];
  const int* dst1 = (const int*)d_in[22];
  const int* gsrc = (const int*)d_in[23];
  const int* gdst = (const int*)d_in[24];
  const int* asrc = (const int*)d_in[25];
  const int* adst = (const int*)d_in[26];
  const int* kp   = (const int*)d_in[27];

  const int F1  = in_sizes[3] / HD;     // 256
  const int F2  = in_sizes[7] / HD;     // 128
  const int N   = in_sizes[0] / F1;     // 30000
  const int E   = in_sizes[19];         // 480000
  const int SIN = in_sizes[2] / N;      // 256

  float* ws = (float*)d_ws;
  size_t o = 0;
  float* h0   = ws + o; o += (size_t)N * HD;
  float* h1   = ws + o; o += (size_t)N * HD;
  float* z0   = ws + o; o += (size_t)N * HD;
  float* z1   = ws + o; o += (size_t)N * HD;
  float* abuf = ws + o; o += (size_t)E * HH;     // attention coeffs (reused per GAT)
  float* Tbuf = ws + o; o += (size_t)N * OUTF;   // proj temp, then GCN accumulator
  float* el   = ws + o; o += (size_t)N * HH;
  float* er   = ws + o; o += (size_t)N * HH;
  float* mb   = ws + o; o += (size_t)N * HH;
  float* sb   = ws + o; o += (size_t)N * HH;
  float* dego = ws + o; o += N;
  float* degi = ws + o; o += N;
  float* dvo  = ws + o; o += N;
  float* dvi  = ws + o; o += N;
  float* wsum = ws + o; o += 2;
  float* beta = ws + o; o += 2;
  float* P    = ws + o; o += (size_t)N * OUTF;
  // reuse of dead regions:
  float* fused = h0;                              // h0 dead after GAT0 scatter
  float* Femb  = h1;                              // h1 dead after GAT1 scatter
  float* xw    = h1 + (size_t)N * OUTF;
  float* p1    = h1 + 2 * (size_t)N * OUTF;
  float* pw    = h1 + 3 * (size_t)N * OUTF;

  float* tr   = (float*)d_out;
  float* Zout = (float*)d_out + 1;

  auto cdiv = [](long a, long b) { return (int)((a + b - 1) / b); };
  auto gemm = [&](const float* A, const float* Bw, const float* bias, float* C,
                  int M, int Ncol, int K) {
    dim3 g(cdiv(M / 16, 16), Ncol / 16);
    k_gemm<<<g, 256, 0, stream>>>(A, Bw, bias, C, M, Ncol, K);
  };

  auto gat = [&](const float* feat, int F, const int* src, const int* dst,
                 const float* W, const float* al, const float* ar, const float* b,
                 float* h, float* z) {
    gemm(feat, W, nullptr, h, N, HD, F);
    k_elr<<<cdiv((long)N * HH, 8), 256, 0, stream>>>(h, al, ar, el, er, N);
    k_fill<<<256, 256, 0, stream>>>(mb, (size_t)N * HH, -__builtin_inff());
    k_fill<<<256, 256, 0, stream>>>(sb, (size_t)N * HH, 0.f);
    k_att_max<<<cdiv((long)E * HH, 256), 256, 0, stream>>>(src, dst, el, er, mb, E);
    k_att_ps<<<cdiv((long)E * HH, 256), 256, 0, stream>>>(src, dst, el, er, mb, sb, abuf, E);
    k_att_norm<<<cdiv((long)E * HH, 256), 256, 0, stream>>>(dst, sb, abuf, E);
    k_fill<<<2048, 256, 0, stream>>>(z, (size_t)N * HD, 0.f);
    k_att_scatter<<<E, 256, 0, stream>>>(src, dst, abuf, h, z, E);
    k_biasclip<<<cdiv((long)N * HD, 256), 256, 0, stream>>>(z, b, (size_t)N * HD);
  };

  // ---- two GAT branches ----
  gat(feat0, F1, src0, dst0, Wgat0, al0, ar0, b0, h0, z0);
  gat(feat1, F2, src1, dst1, Wgat1, al1, ar1, b1, h1, z1);

  // ---- semantic attention fusion ----
  k_fill<<<1, 32, 0, stream>>>(wsum, 2, 0.f);
  gemm(z0, Wp1, bp1, Tbuf, N, OUTF, HD);
  k_tanhdot<<<cdiv(N, 8), 256, 0, stream>>>(Tbuf, wp2, wsum + 0, N);
  gemm(z1, Wp1, bp1, Tbuf, N, OUTF, HD);
  k_tanhdot<<<cdiv(N, 8), 256, 0, stream>>>(Tbuf, wp2, wsum + 1, N);
  k_beta<<<1, 1, 0, stream>>>(wsum, beta, 1.f / (float)N);
  k_fuse<<<cdiv((long)N * HD, 256), 256, 0, stream>>>(z0, z1, beta, fused, (size_t)N * HD);
  gemm(fused, Wf, bf, Femb, N, OUTF, HD);

  // ---- 2-layer GCN on sen_feat ----
  k_fill<<<256, 256, 0, stream>>>(dego, N, 0.f);
  k_fill<<<256, 256, 0, stream>>>(degi, N, 0.f);
  k_deg<<<cdiv(E, 256), 256, 0, stream>>>(gsrc, gdst, dego, degi, E);
  k_rsq<<<cdiv(N, 256), 256, 0, stream>>>(dego, dvo, N);
  k_rsq<<<cdiv(N, 256), 256, 0, stream>>>(degi, dvi, N);
  gemm(sen, Wg1, nullptr, xw, N, OUTF, SIN);
  k_fill<<<2048, 256, 0, stream>>>(Tbuf, (size_t)N * OUTF, 0.f);
  k_gcn_scatter<<<cdiv((long)E * OUTF, 256), 256, 0, stream>>>(gsrc, gdst, xw, dvo, Tbuf, E);
  k_gcnfin<<<cdiv((long)N * OUTF, 256), 256, 0, stream>>>(Tbuf, dvi, bg1, p1, N, 1);
  gemm(p1, Wg2, nullptr, pw, N, OUTF, OUTF);
  k_fill<<<2048, 256, 0, stream>>>(Tbuf, (size_t)N * OUTF, 0.f);
  k_gcn_scatter<<<cdiv((long)E * OUTF, 256), 256, 0, stream>>>(gsrc, gdst, pw, dvo, Tbuf, E);
  k_gcnfin<<<cdiv((long)N * OUTF, 256), 256, 0, stream>>>(Tbuf, dvi, nullptr, P, N, 0);

  // ---- outputs ----
  k_zout<<<cdiv((long)N * OUTF, 256), 256, 0, stream>>>(Femb, P, Zout, (size_t)N * OUTF);
  k_fill<<<1, 32, 0, stream>>>(tr, 1, 0.f);
  k_tr<<<1024, 256, 0, stream>>>(asrc, adst, Zout, kp, tr, E);
}
